// SwinBlock_55250459295904
// MI455X (gfx1250) — matrix operand order repaired and checked by reading, not compiled, not run on hardware
//
#include <hip/hip_runtime.h>
#include <hip/hip_bf16.h>
#include <math.h>

typedef _Float16 v8h  __attribute__((ext_vector_type(8)));
typedef _Float16 v16h __attribute__((ext_vector_type(16)));
typedef float    v8f  __attribute__((ext_vector_type(8)));

#define WSZ   7
#define SSZ   3
#define NHEAD 12
#define DIMC  384
#define HIDC  1536
#define BB    32
#define HH    56
#define WW    56
#define NTOK  (BB * HH * WW)   /* 100352 token rows */
#define NWIN  (BB * 64)        /* 2048 windows      */

// ---------------------------------------------------------------------------
// f32 -> f16 weight conversion
// ---------------------------------------------------------------------------
__global__ void cvt_f32_f16(const float* __restrict__ s, _Float16* __restrict__ d, int n) {
    int i = blockIdx.x * 256 + threadIdx.x;
    if (i < n) d[i] = (_Float16)s[i];
}

// ---------------------------------------------------------------------------
// LayerNorm1 + cyclic shift(-3,-3) + window partition -> xw f16 [2048][49][384]
// one wave (32 lanes) per output row, 8 rows per 256-thread block
// ---------------------------------------------------------------------------
__global__ __launch_bounds__(256)
void ln1_shift_window(const float* __restrict__ x,
                      const float* __restrict__ g,
                      const float* __restrict__ b,
                      _Float16* __restrict__ xw) {
    const int lane = threadIdx.x & 31;
    const long r  = (long)blockIdx.x * 8 + (threadIdx.x >> 5); // shifted-image position
    const int bb  = (int)(r / (HH * WW));
    const int pos = (int)(r % (HH * WW));
    const int hp  = pos / WW, wp = pos % WW;
    const int hs  = (hp + SSZ) % HH, ws2 = (wp + SSZ) % WW;    // roll(-3): h'[i]=x[i+3]
    const float* xr = x + ((long)bb * (HH * WW) + (long)hs * WW + ws2) * DIMC;

    float vals[12];
    float s = 0.f, s2 = 0.f;
#pragma unroll
    for (int i = 0; i < 12; i++) {
        float v = xr[lane + i * 32];
        vals[i] = v; s += v; s2 += v * v;
    }
#pragma unroll
    for (int m = 16; m > 0; m >>= 1) { s += __shfl_xor(s, m); s2 += __shfl_xor(s2, m); }
    const float mu = s * (1.f / DIMC);
    const float rs = rsqrtf(s2 * (1.f / DIMC) - mu * mu + 1e-5f);

    const int win = bb * 64 + (hp / 7) * 8 + (wp / 7);
    const int t   = (hp % 7) * 7 + (wp % 7);
    _Float16* orow = xw + ((long)win * 49 + t) * DIMC;
#pragma unroll
    for (int i = 0; i < 12; i++) {
        int c = lane + i * 32;
        orow[c] = (_Float16)((vals[i] - mu) * rs * g[c] + b[c]);
    }
}

// ---------------------------------------------------------------------------
// Generic WMMA GEMM:  Y[m][n] = sum_k A[m][k] * Wt[n][k]  (+ epilogue)
//   mode 0: +bias, scale q cols (n<384) by 1/sqrt(32), store f16
//   mode 1: +bias, store f32
//   mode 2: +bias, exact GELU, store f16
//   mode 3: +bias, +addend[m*N+n], store f32
// block = 128 threads (4 waves); block tile 64(M) x 64(N); wave tile 16 x 64
// ---------------------------------------------------------------------------
__global__ __launch_bounds__(128)
void gemm_wmma(const _Float16* __restrict__ A,
               const _Float16* __restrict__ Wt,
               const float* __restrict__ bias,
               const float* __restrict__ addend,
               void* __restrict__ outp,
               int N, int K, int mode) {
    const int lane = threadIdx.x & 31;
    const int wave = threadIdx.x >> 5;
    const int lh   = lane & 15;
    const int hi   = lane >> 4;
    const long mb  = (long)blockIdx.x * 64 + wave * 16;
    const int  nb  = blockIdx.y * 64;

    v8f acc[4] = {};
    const _Float16* arow = A + (mb + lh) * (long)K;

    for (int kb = 0; kb < K; kb += 32) {
        // A fragment (16x32 f16): contiguous K = kb+hi*16 .. +15
        v8h a0 = *(const v8h*)(arow + kb + hi * 16);
        v8h a1 = *(const v8h*)(arow + kb + hi * 16 + 8);
        v16h af;
#pragma unroll
        for (int i = 0; i < 8; i++) { af[i] = a0[i]; af[i + 8] = a1[i]; }
#pragma unroll
        for (int t = 0; t < 4; t++) {
            // B fragment: column n = nb+t*16+lh of B  == row n of Wt, contiguous K
            v16h bf = *(const v16h*)(Wt + (long)(nb + t * 16 + lh) * K + kb + hi * 16);
            acc[t] = __builtin_amdgcn_wmma_f32_16x16x32_f16(
                false, af, false, bf, (short)0, acc[t], false, false);
        }
    }

#pragma unroll
    for (int t = 0; t < 4; t++) {
        const int n = nb + t * 16 + lh;
        const float bv = bias ? bias[n] : 0.f;
#pragma unroll
        for (int jj = 0; jj < 8; jj++) {
            const long m = mb + jj + hi * 8;
            float v = acc[t][jj] + bv;
            if (mode == 0) {
                if (n < DIMC) v *= 0.17677669529663687f;   // head_dim^-0.5 on q
                ((_Float16*)outp)[m * N + n] = (_Float16)v;
            } else if (mode == 1) {
                ((float*)outp)[m * N + n] = v;
            } else if (mode == 2) {
                float ge = 0.5f * v * (1.0f + erff(v * 0.70710678118654752f));
                ((_Float16*)outp)[m * N + n] = (_Float16)ge;
            } else {
                ((float*)outp)[m * N + n] = v + addend[m * N + n];
            }
        }
    }
}

// ---------------------------------------------------------------------------
// Windowed attention: one workgroup per (window, head). N=49 padded to 64.
// S = QK^T (WMMA) + rel-pos bias + shift mask; softmax; O = P V (WMMA).
// ---------------------------------------------------------------------------
__global__ __launch_bounds__(128)
void win_attention(const _Float16* __restrict__ qkv,   // [NTOK][1152], q pre-scaled
                   const float* __restrict__ bias_table, // [169][12]
                   _Float16* __restrict__ attn_out) {   // [NTOK][384]
    const int win  = blockIdx.x;
    const int head = blockIdx.y;
    const int tid  = threadIdx.x;
    const int lane = tid & 31, wave = tid >> 5;
    const int lh   = lane & 15, hi = lane >> 4;

    __shared__ __align__(32) _Float16 Qs[64 * 32];
    __shared__ __align__(32) _Float16 Ks[64 * 32];
    __shared__ __align__(32) _Float16 Vt[32 * 64];   // transposed: Vt[d][t]
    __shared__ __align__(32) float    Ssm[64 * 68];
    __shared__ __align__(32) _Float16 Pm[64 * 64];

    // Stage Q, K, V (zero-pad rows 49..63)
    for (int idx = tid; idx < 64 * 32; idx += 128) {
        int t = idx >> 5, d = idx & 31;
        _Float16 qv = (_Float16)0.f, kv = (_Float16)0.f, vv = (_Float16)0.f;
        if (t < 49) {
            long base = ((long)win * 49 + t) * 1152 + head * 32 + d;
            qv = qkv[base];
            kv = qkv[base + 384];
            vv = qkv[base + 768];
        }
        Qs[t * 32 + d] = qv;
        Ks[t * 32 + d] = kv;
        Vt[d * 64 + t] = vv;
    }
    __syncthreads();

    // ---- S = Q K^T : wave handles rows [wave*16, +16), K-dim = 32 (one step)
    const int mb = wave * 16;
    {
        const _Float16* qrow = Qs + (mb + lh) * 32 + hi * 16;
        v8h a0 = *(const v8h*)qrow;
        v8h a1 = *(const v8h*)(qrow + 8);
        v16h af;
#pragma unroll
        for (int i = 0; i < 8; i++) { af[i] = a0[i]; af[i + 8] = a1[i]; }

        v8f sacc[4] = {};
#pragma unroll
        for (int nt = 0; nt < 4; nt++) {
            v16h bf = *(const v16h*)(Ks + (nt * 16 + lh) * 32 + hi * 16);
            sacc[nt] = __builtin_amdgcn_wmma_f32_16x16x32_f16(
                false, af, false, bf, (short)0, sacc[nt], false, false);
        }

        // bias + shift-mask, write to LDS S
        const int wimg = win & 63;
        const int wh = wimg >> 3, wwc = wimg & 7;
#pragma unroll
        for (int nt = 0; nt < 4; nt++) {
            const int j = nt * 16 + lh;
#pragma unroll
            for (int jj = 0; jj < 8; jj++) {
                const int i = mb + jj + hi * 8;
                float sv = -1e30f;
                if (i < 49 && j < 49) {
                    int ri = i / 7, ci = i % 7, rj = j / 7, cj = j % 7;
                    int idx169 = (ri - rj + 6) * 13 + (ci - cj + 6);
                    float bv = bias_table[idx169 * NHEAD + head];
                    int ghi = wh * 7 + ri, gwi = wwc * 7 + ci;
                    int ghj = wh * 7 + rj, gwj = wwc * 7 + cj;
                    int regi = (ghi < 49 ? 0 : (ghi < 53 ? 1 : 2)) * 3
                             + (gwi < 49 ? 0 : (gwi < 53 ? 1 : 2));
                    int regj = (ghj < 49 ? 0 : (ghj < 53 ? 1 : 2)) * 3
                             + (gwj < 49 ? 0 : (gwj < 53 ? 1 : 2));
                    sv = sacc[nt][jj] + bv + (regi != regj ? -100.f : 0.f);
                }
                Ssm[i * 68 + j] = sv;
            }
        }
    }
    __syncthreads();

    // ---- row softmax (64 rows, one thread each), emit P as f16 (zero-padded)
    if (tid < 64) {
        float* srow = Ssm + tid * 68;
        float mx = -1e30f;
        for (int j = 0; j < 49; j++) mx = fmaxf(mx, srow[j]);
        float sum = 0.f;
        for (int j = 0; j < 49; j++) { float e = __expf(srow[j] - mx); srow[j] = e; sum += e; }
        float inv = 1.f / sum;
        _Float16* prow = Pm + tid * 64;
        for (int j = 0; j < 64; j++)
            prow[j] = (j < 49) ? (_Float16)(srow[j] * inv) : (_Float16)0.f;
    }
    __syncthreads();

    // ---- O = P V : K-dim = 64 (two steps), N = 32 (two tiles)
    v8f oacc[2] = {};
#pragma unroll
    for (int kk = 0; kk < 64; kk += 32) {
        const _Float16* pr = Pm + (mb + lh) * 64 + kk + hi * 16;
        v8h p0 = *(const v8h*)pr;
        v8h p1 = *(const v8h*)(pr + 8);
        v16h pf;
#pragma unroll
        for (int i = 0; i < 8; i++) { pf[i] = p0[i]; pf[i + 8] = p1[i]; }
#pragma unroll
        for (int dt = 0; dt < 2; dt++) {
            v16h vf = *(const v16h*)(Vt + (dt * 16 + lh) * 64 + kk + hi * 16);
            oacc[dt] = __builtin_amdgcn_wmma_f32_16x16x32_f16(
                false, pf, false, vf, (short)0, oacc[dt], false, false);
        }
    }
#pragma unroll
    for (int dt = 0; dt < 2; dt++) {
        const int d = dt * 16 + lh;
#pragma unroll
        for (int jj = 0; jj < 8; jj++) {
            const int i = mb + jj + hi * 8;
            if (i < 49)
                attn_out[((long)win * 49 + i) * DIMC + head * 32 + d] = (_Float16)oacc[dt][jj];
        }
    }
}

// ---------------------------------------------------------------------------
// window-reverse + roll(+3,+3) + residual + LayerNorm2
// writes x_new (f32, token layout) and h2 = LN2(x_new) (f16)
// ---------------------------------------------------------------------------
__global__ __launch_bounds__(256)
void residual_ln2(const float* __restrict__ x,
                  const float* __restrict__ proj_out,   // window layout, incl proj_b
                  const float* __restrict__ g2,
                  const float* __restrict__ b2,
                  float* __restrict__ xnew,
                  _Float16* __restrict__ h2) {
    const int lane = threadIdx.x & 31;
    const long r  = (long)blockIdx.x * 8 + (threadIdx.x >> 5); // token index (b, l)
    const int bb  = (int)(r / (HH * WW));
    const int pos = (int)(r % (HH * WW));
    const int hp  = pos / WW, wp = pos % WW;
    const int hs  = (hp - SSZ + HH) % HH, ws2 = (wp - SSZ + WW) % WW;
    const int win = bb * 64 + (hs / 7) * 8 + (ws2 / 7);
    const int t   = (hs % 7) * 7 + (ws2 % 7);
    const float* pr = proj_out + ((long)win * 49 + t) * DIMC;
    const float* xr = x + r * DIMC;

    float vals[12];
    float s = 0.f, s2 = 0.f;
#pragma unroll
    for (int i = 0; i < 12; i++) {
        int c = lane + i * 32;
        float v = xr[c] + pr[c];
        vals[i] = v; s += v; s2 += v * v;
    }
#pragma unroll
    for (int m = 16; m > 0; m >>= 1) { s += __shfl_xor(s, m); s2 += __shfl_xor(s2, m); }
    const float mu = s * (1.f / DIMC);
    const float rs = rsqrtf(s2 * (1.f / DIMC) - mu * mu + 1e-5f);

    float* xo = xnew + r * DIMC;
    _Float16* ho = h2 + r * DIMC;
#pragma unroll
    for (int i = 0; i < 12; i++) {
        int c = lane + i * 32;
        xo[c] = vals[i];
        ho[c] = (_Float16)((vals[i] - mu) * rs * g2[c] + b2[c]);
    }
}

// ---------------------------------------------------------------------------
extern "C" void kernel_launch(void* const* d_in, const int* in_sizes, int n_in,
                              void* d_out, int out_size, void* d_ws, size_t ws_size,
                              hipStream_t stream) {
    const float* x          = (const float*)d_in[0];
    const float* n1g        = (const float*)d_in[3];
    const float* n1b        = (const float*)d_in[4];
    const float* qkv_w      = (const float*)d_in[5];
    const float* qkv_b      = (const float*)d_in[6];
    const float* proj_w     = (const float*)d_in[7];
    const float* proj_b     = (const float*)d_in[8];
    const float* bias_table = (const float*)d_in[9];
    const float* n2g        = (const float*)d_in[10];
    const float* n2b        = (const float*)d_in[11];
    const float* fc1_w      = (const float*)d_in[12];
    const float* fc1_b      = (const float*)d_in[13];
    const float* fc2_w      = (const float*)d_in[14];
    const float* fc2_b      = (const float*)d_in[15];

    char* ws = (char*)d_ws;
    // f16 weights at the front of the workspace
    _Float16* wqkv  = (_Float16*)ws;
    _Float16* wproj = wqkv  + (size_t)1152 * 384;
    _Float16* wfc1  = wproj + (size_t)384 * 384;
    _Float16* wfc2  = wfc1  + (size_t)1536 * 384;

    const size_t o_xw   = (size_t)4 << 20;
    const size_t o_qkv  = o_xw   + (size_t)NTOK * 384 * 2;   // xw f16
    const size_t o_xnew = o_qkv  + (size_t)NTOK * 1152 * 2;  // qkv f16
    const size_t o_h3   = o_xnew + (size_t)NTOK * 384 * 4;   // xnew f32

    _Float16* xw       = (_Float16*)(ws + o_xw);
    _Float16* qkvbuf   = (_Float16*)(ws + o_qkv);
    _Float16* attn_out = xw;                       // xw dead after QKV GEMM
    float*    proj_out = (float*)(ws + o_qkv);     // qkv dead after attention
    float*    xnew     = (float*)(ws + o_xnew);
    _Float16* h2       = xw;                       // attn_out dead after proj GEMM
    _Float16* h3       = (_Float16*)(ws + o_h3);
    float*    out      = (float*)d_out;

    // 1) weights -> f16
    cvt_f32_f16<<<(1152 * 384 + 255) / 256, 256, 0, stream>>>(qkv_w,  wqkv,  1152 * 384);
    cvt_f32_f16<<<(384 * 384  + 255) / 256, 256, 0, stream>>>(proj_w, wproj, 384 * 384);
    cvt_f32_f16<<<(1536 * 384 + 255) / 256, 256, 0, stream>>>(fc1_w,  wfc1,  1536 * 384);
    cvt_f32_f16<<<(384 * 1536 + 255) / 256, 256, 0, stream>>>(fc2_w,  wfc2,  384 * 1536);

    // 2) LN1 + shift + window partition
    ln1_shift_window<<<NTOK / 8, 256, 0, stream>>>(x, n1g, n1b, xw);

    // 3) QKV projection (q scaled by head_dim^-0.5)
    gemm_wmma<<<dim3(NTOK / 64, 1152 / 64), 128, 0, stream>>>(
        xw, wqkv, qkv_b, nullptr, (void*)qkvbuf, 1152, 384, 0);

    // 4) windowed attention per (window, head)
    win_attention<<<dim3(NWIN, NHEAD), 128, 0, stream>>>(qkvbuf, bias_table, attn_out);

    // 5) output projection (+proj_b), f32 window layout
    gemm_wmma<<<dim3(NTOK / 64, 384 / 64), 128, 0, stream>>>(
        attn_out, wproj, proj_b, nullptr, (void*)proj_out, 384, 384, 1);

    // 6) window reverse + unshift + residual + LN2
    residual_ln2<<<NTOK / 8, 256, 0, stream>>>(x, proj_out, n2g, n2b, xnew, h2);

    // 7) MLP fc1 + GELU (f16 out)
    gemm_wmma<<<dim3(NTOK / 64, 1536 / 64), 128, 0, stream>>>(
        h2, wfc1, fc1_b, nullptr, (void*)h3, 1536, 384, 2);

    // 8) MLP fc2 + residual -> final output
    gemm_wmma<<<dim3(NTOK / 64, 384 / 64), 128, 0, stream>>>(
        h3, wfc2, fc2_b, xnew, (void*)out, 384, 1536, 3);
}
